// PowerSpecDens_14843406975602
// MI455X (gfx1250) — compile-verified, matrix operand order checked by reading, not run on hardware
//
#include <hip/hip_runtime.h>
#include <hip/hip_bf16.h>

typedef __attribute__((ext_vector_type(16))) _Float16 v16h;
typedef __attribute__((ext_vector_type(8)))  float    v8f;
typedef __attribute__((ext_vector_type(4)))  _Float16 h4;

#define NS      32768   // samples per row
#define NROWS   2048
#define NFREQS  255     // window length (odd)
#define HOP     127
#define NBINSK  128     // rfft bins
#define NFRAMES 257
#define KSTEPS  8       // 256 padded K / 32

// One block per row. 8 waves; wave w owns bins [16w, 16w+16).
__global__ __launch_bounds__(256) void psd_wmma_kernel(const float* __restrict__ X,
                                                       float* __restrict__ out)
{
    __shared__ _Float16 s_row[NS];   // 64 KB: whole row staged as f16

    const int row    = blockIdx.x;
    const int tid    = threadIdx.x;
    const int lane   = tid & 31;
    const int wave   = tid >> 5;     // N-tile index, 0..7
    const int nlocal = lane & 15;    // column within tile / A-row M
    const int half   = lane >> 4;    // lane half selects K sub-block

    // ---- stage row: f32 global -> f16 LDS (coalesced float4 loads) ----
    const float4* X4 = (const float4*)(X + (size_t)row * NS);
    for (int i = tid; i < NS / 4; i += 256) {
        float4 v = X4[i];
        h4 p;
        p.x = (_Float16)v.x; p.y = (_Float16)v.y;
        p.z = (_Float16)v.z; p.w = (_Float16)v.w;
        *(h4*)&s_row[i * 4] = p;
    }
    __syncthreads();

    // ---- build DFT basis B-fragments (cos, sin) in registers ----
    // 16-bit B 32x16 fragment layout (mirrors A): lanes 0-15 -> N=lane,
    // element e = g*8 + q  (q = jj*2+b) holds K = g*16 + half*8 + q.
    // Per (k,g) chunk the 8 window indices kk are consecutive, so the phase
    // index t = (kk*bin) mod 255 is built incrementally: one modulo per chunk,
    // then t += bin with a single conditional subtract.
    v16h bfrag[2][KSTEPS];
    const int   bin = wave * 16 + nlocal;
    const float w0  = 6.283185307179586f / 255.0f;
    #pragma unroll
    for (int k = 0; k < KSTEPS; ++k) {
        v16h bc, bs;
        #pragma unroll
        for (int g = 0; g < 2; ++g) {
            const int kk0 = k * 32 + g * 16 + half * 8;
            int t = (kk0 * bin) % NFREQS;
            #pragma unroll
            for (int q = 0; q < 8; ++q) {
                const int kk = kk0 + q;            // window index (255 = zero pad)
                const int e  = g * 8 + q;
                float cv = 0.0f, sv = 0.0f;
                if (kk < NFREQS) {
                    __sincosf(w0 * (float)t, &sv, &cv);
                }
                bc[e] = (_Float16)cv;
                bs[e] = (_Float16)sv;
                t += bin;
                if (t >= NFREQS) t -= NFREQS;
            }
        }
        bfrag[0][k] = bc;
        bfrag[1][k] = bs;
    }

    v8f accv = {0.f,0.f,0.f,0.f,0.f,0.f,0.f,0.f};

    // ---- main GEMM loop: M-tiles 0..15 cover frames 0..255, all valid ----
    // 4 independent WMMA chains (cos/sin x K-halves) to keep the XDL pipe full.
    for (int mt = 0; mt < 16; ++mt) {
        v8f c0a = {0.f,0.f,0.f,0.f,0.f,0.f,0.f,0.f};
        v8f c0b = {0.f,0.f,0.f,0.f,0.f,0.f,0.f,0.f};
        v8f c1a = {0.f,0.f,0.f,0.f,0.f,0.f,0.f,0.f};
        v8f c1b = {0.f,0.f,0.f,0.f,0.f,0.f,0.f,0.f};

        const int fbase = (mt * 16 + nlocal) * HOP;   // frame start, always in range

        #pragma unroll
        for (int k = 0; k < KSTEPS; ++k) {
            // 16-bit A 16x32 fragment: lanes 0-15 -> M=lane,
            // element e = g*8 + q holds K = g*16 + half*8 + q
            v16h a;
            #pragma unroll
            for (int e = 0; e < 16; ++e) {
                const int g  = e >> 3;
                const int q  = e & 7;
                const int kl = g * 16 + half * 8 + q;
                a[e] = s_row[fbase + k * 32 + kl];
            }
            if (k < KSTEPS / 2) {
                c0a = __builtin_amdgcn_wmma_f32_16x16x32_f16(false, a, false, bfrag[0][k],
                                                             (short)0, c0a, false, false);
                c1a = __builtin_amdgcn_wmma_f32_16x16x32_f16(false, a, false, bfrag[1][k],
                                                             (short)0, c1a, false, false);
            } else {
                c0b = __builtin_amdgcn_wmma_f32_16x16x32_f16(false, a, false, bfrag[0][k],
                                                             (short)0, c0b, false, false);
                c1b = __builtin_amdgcn_wmma_f32_16x16x32_f16(false, a, false, bfrag[1][k],
                                                             (short)0, c1b, false, false);
            }
        }

        // Re = c0a+c0b, Im = c1a+c1b; vector accumulate |.|^2 (pk_add/pk_fma).
        v8f re = c0a + c0b;
        v8f im = c1a + c1b;
        accv = accv + re * re + im * im;
    }

    // ---- tail M-tile: frames 256..271; only frame 256 (nlocal==0) is real ----
    {
        v8f c0 = {0.f,0.f,0.f,0.f,0.f,0.f,0.f,0.f};
        v8f c1 = {0.f,0.f,0.f,0.f,0.f,0.f,0.f,0.f};

        const bool fok   = (nlocal == 0);
        const int  fbase = 256 * HOP;    // clamp: all lanes read frame 256 (in range)

        #pragma unroll
        for (int k = 0; k < KSTEPS; ++k) {
            v16h a;
            #pragma unroll
            for (int e = 0; e < 16; ++e) {
                const int g  = e >> 3;
                const int q  = e & 7;
                const int kl = g * 16 + half * 8 + q;
                _Float16 v = s_row[fbase + k * 32 + kl];
                a[e] = fok ? v : (_Float16)0.0f;   // select keeps EXEC full for WMMA
            }
            c0 = __builtin_amdgcn_wmma_f32_16x16x32_f16(false, a, false, bfrag[0][k],
                                                        (short)0, c0, false, false);
            c1 = __builtin_amdgcn_wmma_f32_16x16x32_f16(false, a, false, bfrag[1][k],
                                                        (short)0, c1, false, false);
        }

        accv = accv + c0 * c0 + c1 * c1;
    }

    // Horizontal sum of the 8 frame-rows held by this lane...
    float acc = ((accv[0] + accv[1]) + (accv[2] + accv[3]))
              + ((accv[4] + accv[5]) + (accv[6] + accv[7]));
    // ...then lanes L and L+16 hold the same bin: combine halves (wave32).
    acc += __shfl_xor(acc, 16, 32);
    if (lane < 16)
        out[(size_t)row * NBINSK + wave * 16 + lane] = acc * (1.0f / (float)NFRAMES);
}

extern "C" void kernel_launch(void* const* d_in, const int* in_sizes, int n_in,
                              void* d_out, int out_size, void* d_ws, size_t ws_size,
                              hipStream_t stream)
{
    (void)in_sizes; (void)n_in; (void)d_ws; (void)ws_size; (void)out_size;
    const float* X   = (const float*)d_in[0];
    float*       out = (float*)d_out;
    psd_wmma_kernel<<<NROWS, 256, 0, stream>>>(X, out);
}
